// ProgressiveGraphConstruction_15942918603390
// MI455X (gfx1250) — matrix operand order repaired and checked by reading, not compiled, not run on hardware
//
#include <hip/hip_runtime.h>
#include <hip/hip_bf16.h>

typedef __attribute__((ext_vector_type(8)))  float        v8f;
typedef __attribute__((ext_vector_type(16))) __bf16       v16bf;
typedef __attribute__((ext_vector_type(4)))  unsigned int u32x4;
typedef __attribute__((ext_vector_type(2)))  unsigned int u32x2;
typedef __attribute__((ext_vector_type(4)))  float        f32x4;

#define DDIM 768
#define NREG 14
// LDS layout (bytes). Row pads (+16B) keep ds_load_b128 across 16 lanes conflict-free.
#define A_STRIDE   1552                         // 768 bf16 + 16B pad
#define WC_STRIDE  144                          // 64 bf16 + 16B pad
#define WC_BYTES   (128 * WC_STRIDE)            // 18432
#define WC0_OFF    (128 * A_STRIDE)             // 198656
#define WC1_OFF    (WC0_OFF + WC_BYTES)         // 217088  (double buffer)
#define FS_STRIDE  272                          // 128 bf16 + 16B pad
#define FS_STRIP   (16 * FS_STRIDE)             // 4352
#define FS_OFF     (WC1_OFF + WC_BYTES)         // 235520
#define NB_OFF     (FS_OFF + 8 * FS_STRIP)      // 270336
#define SMEM_BYTES (NB_OFF + 8 * 16 * 4)        // 270848  (< 320KB per WGP)

__device__ __forceinline__ unsigned int f2bf(float f) {   // f32 -> bf16 (RNE)
  unsigned int u = __float_as_uint(f);
  u += 0x7FFFu + ((u >> 16) & 1u);
  return u >> 16;
}

union Frag { v16bf v; u32x4 q[2]; };

// Async-DMA one 128x64 bf16 W chunk (16KB) into LDS: 4 b128 transfers per thread.
// ASYNCcnt-tracked; s_wait_xcnt protects the address VGPRs from XNACK-replay reuse.
__device__ __forceinline__ void issue_wc_async(const unsigned short* __restrict__ Wt,
                                               int nbase, int k0, char* wcbase,
                                               int wrow0, int wseg) {
  unsigned long long g0, g1, g2, g3;
  unsigned l0, l1, l2, l3;
  g0 = (unsigned long long)(uintptr_t)(Wt + (size_t)(nbase + wrow0) * DDIM + k0 + wseg * 8);
  g1 = g0 + (unsigned long long)(32 * DDIM * 2);
  g2 = g1 + (unsigned long long)(32 * DDIM * 2);
  g3 = g2 + (unsigned long long)(32 * DDIM * 2);
  l0 = (unsigned)(uintptr_t)(wcbase + wrow0 * WC_STRIDE + wseg * 16);
  l1 = l0 + 32 * WC_STRIDE;
  l2 = l1 + 32 * WC_STRIDE;
  l3 = l2 + 32 * WC_STRIDE;
  asm volatile(
      "global_load_async_to_lds_b128 %0, %4, off\n\t"
      "global_load_async_to_lds_b128 %1, %5, off\n\t"
      "global_load_async_to_lds_b128 %2, %6, off\n\t"
      "global_load_async_to_lds_b128 %3, %7, off\n\t"
      "s_wait_xcnt 0x0"
      :: "v"(l0), "v"(l1), "v"(l2), "v"(l3),
         "v"(g0), "v"(g1), "v"(g2), "v"(g3)
      : "memory");
}

// ---------- prep kernel 1: W[d][e] f32 -> Wt[e][d] bf16 (so B-fragments are contiguous K) ----
__global__ void __launch_bounds__(256) transpose_w_kernel(const float* __restrict__ W,
                                                          unsigned short* __restrict__ Wt) {
  __shared__ float tile[32][33];
  const int tx = threadIdx.x, ty = threadIdx.y;
  const int e0 = blockIdx.x * 32, d0 = blockIdx.y * 32;
#pragma unroll
  for (int i = 0; i < 4; ++i)
    tile[ty + i * 8][tx] = W[(size_t)(d0 + ty + i * 8) * DDIM + e0 + tx];
  __syncthreads();
#pragma unroll
  for (int i = 0; i < 4; ++i)
    Wt[(size_t)(e0 + ty + i * 8) * DDIM + d0 + tx] =
        (unsigned short)f2bf(tile[tx][ty + i * 8]);
}

// ---------- prep kernel 2: S = sum(co_occurrence) ----------
__global__ void __launch_bounds__(256) sum_co_kernel(const float* __restrict__ co,
                                                     float* __restrict__ S) {
  __shared__ float red[256];
  const int t = threadIdx.x;
  red[t] = (t < NREG * NREG) ? co[t] : 0.0f;
  __syncthreads();
  for (int s = 128; s > 0; s >>= 1) {
    if (t < s) red[t] += red[t + s];
    __syncthreads();
  }
  if (t == 0) *S = red[0];
}

// ---------- main fused kernel: 8 batches per workgroup, 256 threads = 8 wave32 ----------
__global__ void __launch_bounds__(256, 1) pgc_main_kernel(
    const float* __restrict__ X,      // [B,14,768]
    const float* __restrict__ area,   // [B,14]
    const float* __restrict__ co,     // [14,14]
    const float* __restrict__ bias,   // [768]
    const unsigned short* __restrict__ Wt,  // [768,768] bf16, Wt[e][d]
    const float* __restrict__ Sp,     // sum(co)
    float* __restrict__ out)          // [B,14,14]
{
  extern __shared__ char smem[];
  const int tid = threadIdx.x;
  const int wave = tid >> 5;
  const int lane = tid & 31;
  const int lane16 = lane & 15;
  const bool hi = lane >= 16;
  const int batch0 = blockIdx.x * 8;

  char* wc[2] = { smem + WC0_OFF, smem + WC1_OFF };
  const int wseg = tid & 7;    // fixed 16B segment per thread
  const int wrow0 = tid >> 3;  // base row; +32 per transfer

  // Kick off chunk 0 of pass 0 immediately: DMA overlaps the whole A-fill below.
  issue_wc_async(Wt, 0, 0, wc[0], wrow0, wseg);

  // ---- Phase 0: zero pad rows (14,15 of each batch), load X -> bf16 into LDS A ----
  for (int i = tid; i < 8 * 2 * (A_STRIDE / 8); i += 256) {
    const int row = i / (A_STRIDE / 8);           // 0..15 -> (batch g, pad row rr)
    const int g = row >> 1, rr = row & 1;
    const int off = (i % (A_STRIDE / 8)) * 8;
    *(unsigned long long*)(smem + (size_t)(g * 16 + 14 + rr) * A_STRIDE + off) = 0ull;
  }
  for (int i = tid; i < 8 * NREG * (DDIM / 4); i += 256) {
    const int c4 = i % (DDIM / 4);
    const int gr = i / (DDIM / 4);
    const int g = gr / NREG, r = gr % NREG;
    const f32x4 f = *(const f32x4*)(X + ((size_t)(batch0 + g) * NREG + r) * DDIM + c4 * 4);
    u32x2 p;
    p.x = f2bf(f.x) | (f2bf(f.y) << 16);
    p.y = f2bf(f.z) | (f2bf(f.w) << 16);
    *(u32x2*)(smem + (size_t)(g * 16 + r) * A_STRIDE + c4 * 8) = p;
  }

  const float cwinv = 1.0f / (*Sp + 1e-8f);

  // wave job: 2 M-strips x 4 N-tiles (balances A/B LDS fragment traffic)
  const int sp = wave & 3;   // strip pair -> strips 2sp, 2sp+1
  const int nh = wave >> 2;  // n-half of the 128-col pass

  v8f G = {0, 0, 0, 0, 0, 0, 0, 0};  // Gram accumulator for batch == wave

  for (int p = 0; p < 6; ++p) {      // 6 passes of 128 output columns
    const int nbase = p * 128;
    v8f acc[2][4];
#pragma unroll
    for (int ss = 0; ss < 2; ++ss)
#pragma unroll
      for (int j = 0; j < 4; ++j) { v8f z = {0,0,0,0,0,0,0,0}; acc[ss][j] = z; }

    for (int c = 0; c < 12; ++c) {   // K chunks of 64, double-buffered async DMA
      const int k0 = c * 64;
      __syncthreads();   // (A) all waves finished reading the buffer we now overwrite
      if (c < 11) {
        issue_wc_async(Wt, nbase, k0 + 64, wc[(c + 1) & 1], wrow0, wseg);
        asm volatile("s_wait_asynccnt 0x4" ::: "memory");  // chunk c retired (in-order)
      } else {
        asm volatile("s_wait_asynccnt 0x0" ::: "memory");
      }
      __syncthreads();   // (B) chunk c visible to all waves
      const char* wcbuf = wc[c & 1];
#pragma unroll
      for (int kk = 0; kk < 2; ++kk) {          // two K=32 WMMAs per chunk
        const int kb = kk * 32;
        Frag a[2], bfr[4];
#pragma unroll
        for (int ss = 0; ss < 2; ++ss) {        // A fragment: lane16 = M row
          const int arow = (sp * 2 + ss) * 16 + lane16;
          const char* pa = smem + (size_t)arow * A_STRIDE + 2 * (k0 + kb) + (hi ? 16 : 0);
          a[ss].q[0] = *(const u32x4*)(pa);
          a[ss].q[1] = *(const u32x4*)(pa + 32);
        }
#pragma unroll
        for (int j = 0; j < 4; ++j) {           // B fragment: lane16 = N col, contiguous K
          const int nrow = nh * 64 + j * 16 + lane16;
          const char* pb = wcbuf + nrow * WC_STRIDE + 2 * kb + (hi ? 16 : 0);
          bfr[j].q[0] = *(const u32x4*)(pb);
          bfr[j].q[1] = *(const u32x4*)(pb + 32);
        }
#pragma unroll
        for (int ss = 0; ss < 2; ++ss)
#pragma unroll
          for (int j = 0; j < 4; ++j)
            acc[ss][j] = __builtin_amdgcn_wmma_f32_16x16x32_bf16(
                false, a[ss].v, false, bfr[j].v, (short)0, acc[ss][j], false, false);
      }
    }

    // bias add + spill F-pass (bf16) to LDS, laid out per strip (= per batch)
#pragma unroll
    for (int ss = 0; ss < 2; ++ss) {
      const int strip = sp * 2 + ss;
#pragma unroll
      for (int j = 0; j < 4; ++j) {
        const int ncol = nh * 64 + j * 16 + lane16;    // column within pass
        const float bv = bias[nbase + ncol];
#pragma unroll
        for (int v = 0; v < 8; ++v) {
          const int m = v + (hi ? 8 : 0);              // row within strip
          const float fv = acc[ss][j][v] + bv;
          *(unsigned short*)(smem + FS_OFF + strip * FS_STRIP + m * FS_STRIDE + ncol * 2) =
              (unsigned short)f2bf(fv);
        }
      }
    }
    __syncthreads();

    // Prefetch chunk 0 of the next pass: DMA overlaps the Gram WMMAs below.
    if (p < 5) issue_wc_async(Wt, nbase + 128, 0, wc[0], wrow0, wseg);

    // Gram via WMMA: wave w accumulates G += F_w(16x128) @ F_w^T ; norms = diagonal
#pragma unroll
    for (int kc = 0; kc < 4; ++kc) {
      const int kb = kc * 32;
      const char* prow = smem + FS_OFF + wave * FS_STRIP + lane16 * FS_STRIDE + 2 * kb;
      Frag ga, gb;
      const char* pa = prow + (hi ? 16 : 0);           // A layout K slices {0-7,16-23}/{8-15,24-31}
      ga.q[0] = *(const u32x4*)(pa);
      ga.q[1] = *(const u32x4*)(pa + 32);
      const char* pb = prow + (hi ? 16 : 0);           // B layout K slices {0-7,16-23}/{8-15,24-31}
      gb.q[0] = *(const u32x4*)(pb);
      gb.q[1] = *(const u32x4*)(pb + 32);
      G = __builtin_amdgcn_wmma_f32_16x16x32_bf16(false, ga.v, false, gb.v, (short)0, G,
                                                  false, false);
    }
  }

  // ---- Epilogue (per wave == per batch): cosine sim, area & co weights, softmax over m ----
  // C layout: element [m][n]: vgpr = m&7, half = m>=8, lane16 = n. Diagonal -> norms.
  float* nb = (float*)(smem + NB_OFF) + wave * 16;
  {
    float dval = 0.0f;
#pragma unroll
    for (int v = 0; v < 8; ++v) {
      const int m = v + (hi ? 8 : 0);
      if (m == lane16) dval = G[v];
    }
    if ((!hi && lane16 < 8) || (hi && lane16 >= 8)) nb[lane16] = dval;  // same-wave LDS order
  }
  const int bidx = batch0 + wave;
  const int n = lane16;                          // output row
  const float invn = 1.0f / fmaxf(sqrtf(fmaxf(nb[n], 0.0f)), 1e-12f);

  const float* ar = area + (size_t)bidx * NREG;  // wave-uniform row -> scalarized loads
  float amax = 0.0f;
#pragma unroll
  for (int i = 0; i < NREG; ++i) amax = fmaxf(amax, ar[i]);
  const float awinv = 1.0f / (amax * amax + 1e-8f);
  const float an = (n < NREG) ? ar[n] : 0.0f;

  float e[8];
  float lmax = -3.4e38f;
#pragma unroll
  for (int v = 0; v < 8; ++v) {
    const int m = v + (hi ? 8 : 0);
    float val = -3.4e38f;
    if (m < NREG && n < NREG) {
      const float invm = 1.0f / fmaxf(sqrtf(fmaxf(nb[m], 0.0f)), 1e-12f);
      const float sim = G[v] * invn * invm;
      const float aw = an * ar[m] * awinv;
      const float cw = co[n * NREG + m] * cwinv;
      val = sim * aw * cw;
    }
    e[v] = val;
    lmax = fmaxf(lmax, val);
  }
  const float rmax = fmaxf(lmax, __shfl_xor(lmax, 16, 32));  // join half-rows (m 0-7 | 8-13)
  float lsum = 0.0f;
#pragma unroll
  for (int v = 0; v < 8; ++v) {
    const int m = v + (hi ? 8 : 0);
    if (m < NREG && n < NREG) { e[v] = __expf(e[v] - rmax); lsum += e[v]; }
    else e[v] = 0.0f;
  }
  const float rinv = 1.0f / (lsum + __shfl_xor(lsum, 16, 32));
  if (n < NREG) {
    float* orow = out + (size_t)bidx * (NREG * NREG) + n * NREG;
#pragma unroll
    for (int v = 0; v < 8; ++v) {
      const int m = v + (hi ? 8 : 0);
      if (m < NREG) orow[m] = e[v] * rinv;
    }
  }
}

extern "C" void kernel_launch(void* const* d_in, const int* in_sizes, int n_in,
                              void* d_out, int out_size, void* d_ws, size_t ws_size,
                              hipStream_t stream) {
  const float* X    = (const float*)d_in[0];   // region_features [B,14,768]
  const float* area = (const float*)d_in[1];   // [B,14]
  const float* co   = (const float*)d_in[2];   // [14,14]
  const float* W    = (const float*)d_in[3];   // [768,768]
  const float* b    = (const float*)d_in[4];   // [768]
  float* out = (float*)d_out;

  const int B = in_sizes[0] / (NREG * DDIM);   // 8192

  unsigned short* Wt = (unsigned short*)d_ws;                       // 768*768*2 = 1.18 MB
  float* Sp = (float*)((char*)d_ws + (size_t)DDIM * DDIM * 2);      // + 4 B

  (void)ws_size; (void)n_in; (void)out_size;
  hipFuncSetAttribute(reinterpret_cast<const void*>(pgc_main_kernel),
                      hipFuncAttributeMaxDynamicSharedMemorySize, SMEM_BYTES);

  dim3 tg(DDIM / 32, DDIM / 32), tb(32, 8);
  transpose_w_kernel<<<tg, tb, 0, stream>>>(W, Wt);
  sum_co_kernel<<<1, 256, 0, stream>>>(co, Sp);
  pgc_main_kernel<<<B / 8, 256, SMEM_BYTES, stream>>>(X, area, co, b, Wt, Sp, out);
}
